// GNTO_QF_Profile_70196945485987
// MI455X (gfx1250) — matrix operand order repaired and checked by reading, not compiled, not run on hardware
//
#include <hip/hip_runtime.h>
#include <hip/hip_bf16.h>
#include <math.h>

// ---------------- problem constants (match reference) ----------------
#define NN      100000      // nodes
#define EE      1600000     // edges
#define GG      128         // graphs
#define IN_DIM  329
#define KPAD    352         // 11 * 32, zero-padded K
#define HH      4           // heads
#define CC      64          // channels per head
#define HC      (HH*CC)     // 256
#define EN      (EE+NN)     // edges + self loops
#define NEG_SLOPE 0.2f

typedef __attribute__((ext_vector_type(16))) __bf16 v16bf;
typedef __attribute__((ext_vector_type(8)))  __bf16 v8bf;
typedef __attribute__((ext_vector_type(8)))  float  v8f;

// ---------------- utility ----------------
__global__ void fill_f32(float* __restrict__ p, long n, float v) {
    long i = (long)blockIdx.x * blockDim.x + threadIdx.x;
    long stride = (long)gridDim.x * blockDim.x;
    for (; i < n; i += stride) p[i] = v;
}

// Ordered-int float atomic max (valid with -inf init; mixed signs handled)
__device__ __forceinline__ void atomic_max_f32(float* addr, float val) {
    if (val >= 0.0f) atomicMax((int*)addr, __float_as_int(val));
    else             atomicMin((unsigned int*)addr, __float_as_uint(val));
}

// ---------------- 0) pre-swizzle W -> bf16 fragment order ----------------
// Bsw[col][k] (KPAD padded, zero tail): a lane's B fragment for K-step k0 is
// the 8 bf16 at Bsw[col*KPAD + k0 + half*8] plus the 8 at +16  (two 16B loads).
__global__ void prep_B_kernel(const float* __restrict__ W, __bf16* __restrict__ Bsw) {
    int t = blockIdx.x * blockDim.x + threadIdx.x;
    if (t >= HC * KPAD) return;
    int col = t / KPAD, k = t - col * KPAD;
    Bsw[t] = (__bf16)(k < IN_DIM ? W[(size_t)k * HC + col] : 0.0f);
}

// ---------------- 1) xp = x @ W  via bf16 WMMA ----------------
// grid = NN/16 blocks of 512 threads (16 waves). Block computes the full
// 16x256 output strip; wave w owns columns w*16..w*16+15. A tile staged to
// LDS once (x bytes cross HBM exactly once); B fragments read from the
// L2-resident pre-swizzled Bsw with aligned b128 loads. No syncs in K loop.
__global__ __launch_bounds__(512)
void gemm_xp_wmma(const float* __restrict__ x, const __bf16* __restrict__ Bsw,
                  float* __restrict__ xp) {
    __shared__ __bf16 ldsA[16 * KPAD];     // 11.0 KB
    const int tid  = threadIdx.x;
    const int wave = tid >> 5;             // 0..15
    const int lane = tid & 31;
    const int half = lane >> 4;            // 0: lanes 0-15, 1: lanes 16-31
    const int l    = lane & 15;
    const int mbase = blockIdx.x * 16;

    // ---- stage A tile (16 x KPAD bf16, zero-padded): wave w loads row w ----
    {
        const int r = tid >> 5;            // one wave per row
        const float* xr = x + (size_t)(mbase + r) * IN_DIM;
        for (int k = lane; k < KPAD; k += 32) {        // coalesced 128B lines
            float v = (k < IN_DIM) ? xr[k] : 0.0f;
            ldsA[r * KPAD + k] = (__bf16)v;
        }
    }
    __syncthreads();

    const int col = wave * 16 + l;
    const __bf16* bp = Bsw + (size_t)col * KPAD;       // 16B-aligned (704B rows)

    v8f c = {};
    #pragma unroll 2
    for (int k0 = 0; k0 < KPAD; k0 += 32) {
        // A 16x32 frag: half 0 -> K {0..7,16..23}+k0 of row l; half 1 -> +8
        v8bf aLo = *(const v8bf*)&ldsA[l * KPAD + k0 + half * 8];
        v8bf aHi = *(const v8bf*)&ldsA[l * KPAD + k0 + 16 + half * 8];
        // B 32x16 frag: lane = col; half 0 -> K {0..7,16..23}+k0; half 1 -> +8
        v8bf bLo = *(const v8bf*)&bp[k0 + half * 8];
        v8bf bHi = *(const v8bf*)&bp[k0 + 16 + half * 8];

        v16bf a = __builtin_shufflevector(aLo, aHi,
                    0,1,2,3,4,5,6,7, 8,9,10,11,12,13,14,15);
        v16bf b = __builtin_shufflevector(bLo, bHi,
                    0,1,2,3,4,5,6,7, 8,9,10,11,12,13,14,15);

        c = __builtin_amdgcn_wmma_f32_16x16x32_bf16(
                false, a, false, b, (short)0, c, false, false);
    }

    // C/D layout: VGPR r -> row mbase + r + half*8, col
    #pragma unroll
    for (int r = 0; r < 8; ++r)
        xp[(size_t)(mbase + r + half * 8) * HC + col] = c[r];
}

// ---------------- 2) attention logits per node ----------------
__global__ void att_kernel(const float* __restrict__ xp,
                           const float* __restrict__ att_src,
                           const float* __restrict__ att_dst,
                           float* __restrict__ a_src, float* __restrict__ a_dst) {
    int t = blockIdx.x * blockDim.x + threadIdx.x;
    if (t >= NN * HH) return;
    int n = t >> 2, h = t & 3;
    const float* row = xp + (size_t)n * HC + h * CC;
    float s = 0.f, d = 0.f;
    #pragma unroll 8
    for (int c = 0; c < CC; ++c) {
        float v = row[c];
        s = fmaf(v, att_src[h * CC + c], s);
        d = fmaf(v, att_dst[h * CC + c], d);
    }
    a_src[t] = s; a_dst[t] = d;
}

__device__ __forceinline__ void edge_endpoints(int e, const int* __restrict__ ei,
                                               int* src, int* dst) {
    if (e < EE) { *src = ei[e]; *dst = ei[EE + e]; }
    else        { *src = e - EE; *dst = e - EE; }      // self loop
}

__device__ __forceinline__ float leaky(float v) {
    return v > 0.0f ? v : NEG_SLOPE * v;
}

// ---------------- 3a) segment max of logits over dst ----------------
__global__ void edge_max_kernel(const int* __restrict__ ei,
                                const float* __restrict__ a_src,
                                const float* __restrict__ a_dst,
                                float* __restrict__ mx) {
    int t = blockIdx.x * blockDim.x + threadIdx.x;
    if (t >= EN * HH) return;
    int e = t >> 2, h = t & 3;
    int s, d; edge_endpoints(e, ei, &s, &d);
    float logit = leaky(a_src[s * HH + h] + a_dst[d * HH + h]);
    atomic_max_f32(&mx[d * HH + h], logit);
}

// ---------------- 3b) segment sum of exp ----------------
__global__ void edge_sum_kernel(const int* __restrict__ ei,
                                const float* __restrict__ a_src,
                                const float* __restrict__ a_dst,
                                const float* __restrict__ mx,
                                float* __restrict__ denom) {
    int t = blockIdx.x * blockDim.x + threadIdx.x;
    if (t >= EN * HH) return;
    int e = t >> 2, h = t & 3;
    int s, d; edge_endpoints(e, ei, &s, &d);
    float logit = leaky(a_src[s * HH + h] + a_dst[d * HH + h]);
    float ex = __expf(logit - mx[d * HH + h]);
    atomicAdd(&denom[d * HH + h], ex);
}

// ---------------- 3c) weighted aggregation ----------------
__global__ void edge_agg_kernel(const int* __restrict__ ei,
                                const float* __restrict__ a_src,
                                const float* __restrict__ a_dst,
                                const float* __restrict__ mx,
                                const float* __restrict__ denom,
                                const float* __restrict__ xp,
                                float* __restrict__ agg) {
    int t = blockIdx.x * blockDim.x + threadIdx.x;
    if (t >= EN * HH) return;
    int e = t >> 2, h = t & 3;
    int s, d; edge_endpoints(e, ei, &s, &d);
    float logit = leaky(a_src[s * HH + h] + a_dst[d * HH + h]);
    float alpha = __expf(logit - mx[d * HH + h]) / denom[d * HH + h];
    const float* xr = xp + (size_t)s * HC + h * CC;
    float* ar = agg + (size_t)d * HC + h * CC;
    #pragma unroll 4
    for (int c = 0; c < CC; ++c)
        atomicAdd(&ar[c], xr[c] * alpha);
}

// ---------------- 4) head-mean + relu + graph pooling ----------------
__global__ void node_pool_kernel(const float* __restrict__ agg,
                                 const float* __restrict__ bias,
                                 const int* __restrict__ batch,
                                 float* __restrict__ pooled,
                                 float* __restrict__ counts) {
    int t = blockIdx.x * blockDim.x + threadIdx.x;
    if (t >= NN * CC) return;
    int n = t >> 6, c = t & 63;
    const float* ar = agg + (size_t)n * HC;
    float v = 0.25f * (ar[c] + ar[CC + c] + ar[2 * CC + c] + ar[3 * CC + c]) + bias[c];
    v = fmaxf(v, 0.0f);
    int g = batch[n];
    atomicAdd(&pooled[g * CC + c], v);
    if (c == 0) atomicAdd(&counts[g], 1.0f);
}

// ---------------- 5) head MLP ----------------
__global__ void mlp1_kernel(const float* __restrict__ pooled,
                            const float* __restrict__ counts,
                            const float* __restrict__ w1,
                            const float* __restrict__ b1,
                            float* __restrict__ hidden) {
    int t = blockIdx.x * blockDim.x + threadIdx.x;
    if (t >= GG * 64) return;
    int g = t >> 6, j = t & 63;
    float inv = 1.0f / fmaxf(counts[g], 1.0f);
    float acc = b1[j];
    #pragma unroll 8
    for (int c = 0; c < CC; ++c)
        acc = fmaf(pooled[g * CC + c] * inv, w1[c * 64 + j], acc);
    hidden[t] = fmaxf(acc, 0.0f);
}

__global__ void mlp2_kernel(const float* __restrict__ hidden,
                            const float* __restrict__ w2,
                            const float* __restrict__ b2,
                            float* __restrict__ y) {
    int g = blockIdx.x * blockDim.x + threadIdx.x;
    if (g >= GG) return;
    float acc = b2[0];
    #pragma unroll 8
    for (int j = 0; j < 64; ++j)
        acc = fmaf(hidden[g * 64 + j], w2[j], acc);
    y[g] = 1.0f / (1.0f + __expf(-acc));
}

// ---------------- launcher ----------------
extern "C" void kernel_launch(void* const* d_in, const int* in_sizes, int n_in,
                              void* d_out, int out_size, void* d_ws, size_t ws_size,
                              hipStream_t stream) {
    const float* x       = (const float*)d_in[0];
    const float* W       = (const float*)d_in[1];
    const float* att_src = (const float*)d_in[2];
    const float* att_dst = (const float*)d_in[3];
    const float* bias    = (const float*)d_in[4];
    const float* w1      = (const float*)d_in[5];
    const float* b1      = (const float*)d_in[6];
    const float* w2      = (const float*)d_in[7];
    const float* b2      = (const float*)d_in[8];
    const int*   ei      = (const int*)d_in[9];
    const int*   batch   = (const int*)d_in[10];
    float* y = (float*)d_out;

    // carve workspace (all offsets 256B aligned)
    char* ws = (char*)d_ws;
    size_t off = 0;
    auto take = [&](size_t bytes) -> void* {
        char* p = ws + off; off = (off + bytes + 255) & ~(size_t)255; return (void*)p; };
    float*  xp     = (float*) take((size_t)NN * HC * sizeof(float));   // 102.4 MB
    float*  a_src  = (float*) take((size_t)NN * HH * sizeof(float));
    float*  a_dst  = (float*) take((size_t)NN * HH * sizeof(float));
    float*  mx     = (float*) take((size_t)NN * HH * sizeof(float));
    float*  denom  = (float*) take((size_t)NN * HH * sizeof(float));
    float*  agg    = (float*) take((size_t)NN * HC * sizeof(float));   // 102.4 MB
    float*  pooled = (float*) take((size_t)GG * CC * sizeof(float));
    float*  counts = (float*) take((size_t)GG * sizeof(float));
    float*  hidden = (float*) take((size_t)GG * 64 * sizeof(float));
    __bf16* Bsw    = (__bf16*)take((size_t)HC * KPAD * sizeof(__bf16)); // 176 KB
    (void)ws_size; (void)in_sizes; (void)n_in; (void)out_size;

    // deterministic init of all accumulation buffers (every call)
    fill_f32<<<2048, 256, 0, stream>>>(agg,    (long)NN * HC, 0.0f);
    fill_f32<<<256,  256, 0, stream>>>(denom,  (long)NN * HH, 0.0f);
    fill_f32<<<256,  256, 0, stream>>>(mx,     (long)NN * HH, -INFINITY);
    fill_f32<<<32,   256, 0, stream>>>(pooled, (long)GG * CC, 0.0f);
    fill_f32<<<1,    256, 0, stream>>>(counts, (long)GG, 0.0f);

    // 0) swizzle W into bf16 fragment order (L2-resident afterwards)
    prep_B_kernel<<<(HC * KPAD + 255) / 256, 256, 0, stream>>>(W, Bsw);

    // 1) big GEMM on the matrix pipes
    gemm_xp_wmma<<<NN / 16, 512, 0, stream>>>(x, Bsw, xp);

    // 2) per-node attention logits
    att_kernel<<<(NN * HH + 255) / 256, 256, 0, stream>>>(xp, att_src, att_dst, a_src, a_dst);

    // 3) edge softmax + aggregation (bandwidth/atomic bound)
    int eb = (EN * HH + 255) / 256;
    edge_max_kernel<<<eb, 256, 0, stream>>>(ei, a_src, a_dst, mx);
    edge_sum_kernel<<<eb, 256, 0, stream>>>(ei, a_src, a_dst, mx, denom);
    edge_agg_kernel<<<eb, 256, 0, stream>>>(ei, a_src, a_dst, mx, denom, xp, agg);

    // 4) head mean + relu + graph mean pool
    node_pool_kernel<<<(NN * CC + 255) / 256, 256, 0, stream>>>(agg, bias, batch, pooled, counts);

    // 5) tiny head MLP
    mlp1_kernel<<<(GG * 64 + 255) / 256, 256, 0, stream>>>(pooled, counts, w1, b1, hidden);
    mlp2_kernel<<<1, 128, 0, stream>>>(hidden, w2, b2, y);
}
